// VarConv2d_25967372272299
// MI455X (gfx1250) — matrix operand order repaired and checked by reading, not compiled, not run on hardware
//
#include <hip/hip_runtime.h>

// ---------------------------------------------------------------------------
// VarConv2d (variational conv head) for MI455X / gfx1250, wave32 + WMMA bf16.
//   x      [16,64,128,128] f32
//   w_mu   [128,576] f32   b_mu   [128] f32
//   w_lv   [128,576] f32   b_lv   [128] f32
//   out    [16,128,128,128] f32  ++  varloss scalar (tail of d_out)
//
// Dual GEMM (mu / logvar): M=O=128, N=B*H*W=262144, K=C*9=576,
// via v_wmma_f32_16x16x32_bf16.
//  - prologue kernel converts weights fp32 -> bf16 into d_ws (L2-resident)
//  - main kernel: A frags read straight from global (2x b128 per frag),
//    im2col B tile (128 N x 32 K) double-buffered in LDS, 1 barrier/K-step,
//    wave tile = 16M x 128N -> 16 WMMA per K-step per wave.
// ---------------------------------------------------------------------------

#define BIMG   16
#define CIN    64
#define HH     128
#define WW     128
#define OOUT   128
#define KTOT   576              // CIN * 9
#define KSTEP  32
#define NSTEPS (KTOT / KSTEP)   // 18
#define NTILE  128
#define HWSZ   (HH * WW)
#define LDP    40               // Bs row stride in halves: 80B, 16B-aligned
#define BUFH   (NTILE * LDP)    // 5120 halves per LDS buffer
#define WELEM  (OOUT * KTOT)    // 73728 elements per weight tensor

typedef __attribute__((ext_vector_type(16))) __bf16 v16bf;
typedef __attribute__((ext_vector_type(8)))  float  v8f;

union Frag { uint4 q[2]; v16bf v; };

__device__ __forceinline__ unsigned short f2bf(float f) {
    unsigned u = __float_as_uint(f);
    u += 0x7FFFu + ((u >> 16) & 1u);        // round-to-nearest-even
    return (unsigned short)(u >> 16);
}

__device__ __forceinline__ unsigned pcg(unsigned v) {
    unsigned s = v * 747796405u + 2891336453u;
    unsigned w = ((s >> ((s >> 28) + 4u)) ^ s) * 277803737u;
    return (w >> 22) ^ w;
}

// prologue: weights fp32 -> bf16 into d_ws, and zero the loss scalar
__global__ __launch_bounds__(256)
void convert_weights_kernel(const float* __restrict__ wmu,
                            const float* __restrict__ wlv,
                            unsigned short* __restrict__ wbf,
                            float* __restrict__ loss)
{
    int idx = blockIdx.x * 256 + threadIdx.x;
    if (idx == 0) *loss = 0.0f;
    if (idx < WELEM) {
        wbf[idx]         = f2bf(wmu[idx]);
        wbf[WELEM + idx] = f2bf(wlv[idx]);
    }
}

__global__ __launch_bounds__(256)
void varconv_wmma_kernel(const float* __restrict__ x,
                         const unsigned short* __restrict__ wbf,
                         const float* __restrict__ bmu,
                         const float* __restrict__ blv,
                         float* __restrict__ out,
                         float* __restrict__ loss)
{
    __shared__ unsigned short Bs[2 * BUFH];      // double-buffered im2col tile
    __shared__ float sbm[OOUT], sbl[OOUT];       // biases

    const int tid  = threadIdx.x;
    const int bimg = blockIdx.y;
    const int s0   = blockIdx.x * NTILE;

    const int wv   = tid >> 5;                   // wave id: M stripe (16 rows)
    const int lane = tid & 31;
    const int hi   = lane >> 4;
    const int lo   = lane & 15;

    if (tid < OOUT) sbm[tid] = bmu[tid];
    else            sbl[tid - OOUT] = blv[tid - OOUT];

    // ---- staging invariants: this thread always fills column n_st ----------
    const int n_st  = tid & 127;                 // column within N tile
    const int kb_st = tid >> 7;                  // 0/1: kk = kb_st + 2e
    const int sspat = s0 + n_st;
    const int y0    = sspat >> 7;
    const int x0    = sspat & 127;
    const float* __restrict__ xb = x + (size_t)bimg * CIN * HWSZ;

    const unsigned short* __restrict__ wa = wbf;          // mu weights (bf16)
    const unsigned short* __restrict__ wl = wbf + WELEM;  // logvar weights

    // A fragment: lane holds M = wv*16+lo; halves 0..7 -> K=8*hi..,
    // halves 8..15 -> K=16+8*hi.. (ISA 7.12.2, 16-bit A 16x32)
    const int abase = (wv * 16 + lo) * KTOT;

    v8f accm[8], accl[8];
    #pragma unroll
    for (int i = 0; i < 8; ++i)
        #pragma unroll
        for (int j = 0; j < 8; ++j) { accm[i][j] = 0.0f; accl[i][j] = 0.0f; }

    // im2col stage of K-slice kc into LDS buffer `buf`
    auto stage = [&](int kc, int buf) {
        unsigned short* dst = &Bs[buf * BUFH + n_st * LDP];
        const int k0 = kc * KSTEP;
        #pragma unroll
        for (int e = 0; e < 16; ++e) {
            int kk = kb_st + 2 * e;
            int kg = k0 + kk;
            int c  = kg / 9;
            int r  = kg - 9 * c;
            int di = r / 3 - 1;
            int dj = r - 3 * (r / 3) - 1;
            int yy = y0 + di;
            int xx = x0 + dj;
            float v = 0.0f;
            if ((unsigned)yy < (unsigned)HH && (unsigned)xx < (unsigned)WW)
                v = xb[(c * HH + yy) * WW + xx];
            dst[kk] = f2bf(v);
        }
    };

    stage(0, 0);
    __syncthreads();

    for (int kc = 0; kc < NSTEPS; ++kc) {
        const int cur = kc & 1;
        if (kc + 1 < NSTEPS)
            stage(kc + 1, 1 - cur);              // overlaps with WMMA below

        const int k0 = kc * KSTEP;
        Frag fa, fl;
        fa.q[0] = *(const uint4*)&wa[abase + k0 + 8 * hi];
        fa.q[1] = *(const uint4*)&wa[abase + k0 + 8 * hi + 16];
        fl.q[0] = *(const uint4*)&wl[abase + k0 + 8 * hi];
        fl.q[1] = *(const uint4*)&wl[abase + k0 + 8 * hi + 16];

        const unsigned short* bsrc = &Bs[cur * BUFH];
        #pragma unroll
        for (int nt = 0; nt < 8; ++nt) {
            // B frag: lane holds N = nt*16+lo, K = 8*hi..8*hi+7, 16+8*hi..16+8*hi+7
            Frag fb;
            const int ncol = nt * 16 + lo;
            fb.q[0] = *(const uint4*)&bsrc[ncol * LDP + 8 * hi];
            fb.q[1] = *(const uint4*)&bsrc[ncol * LDP + 8 * hi + 16];

            accm[nt] = __builtin_amdgcn_wmma_f32_16x16x32_bf16(
                false, fa.v, false, fb.v, (short)0, accm[nt], false, false);
            accl[nt] = __builtin_amdgcn_wmma_f32_16x16x32_bf16(
                false, fl.v, false, fb.v, (short)0, accl[nt], false, false);
        }
        __syncthreads();                         // one barrier per K-step
    }

    // ---- epilogue: bias, reparameterization (shared Box-Muller pair), KLD --
    float part = 0.0f;

    #pragma unroll
    for (int nt = 0; nt < 8; ++nt) {
        const int n = s0 + nt * 16 + lo;
        #pragma unroll
        for (int r = 0; r < 8; r += 2) {
            const int o0 = wv * 16 + r + 8 * hi;     // C/D layout: M = r + 8*hi

            float mu0 = accm[nt][r]     + sbm[o0];
            float lv0 = accl[nt][r]     + sbl[o0];
            float mu1 = accm[nt][r + 1] + sbm[o0 + 1];
            float lv1 = accl[nt][r + 1] + sbl[o0 + 1];

            unsigned g0 = (unsigned)((bimg * OOUT + o0) * HWSZ + n);
            unsigned g1 = g0 + HWSZ;                 // o1 = o0+1, same n

            unsigned h1 = pcg(g0);
            unsigned h2 = pcg(g0 ^ 0x9E3779B9u);
            float u1 = (float)((h1 >> 8) + 1u) * (1.0f / 16777216.0f);
            float u2 = (float)(h2 >> 8) * (1.0f / 16777216.0f);
            float rad   = __builtin_sqrtf(-2.0f * __logf(u1));
            float theta = 6.2831853f * u2;

            float std0 = __expf(0.5f * lv0);         // exp(lv) = std*std
            float std1 = __expf(0.5f * lv1);
            float ov0  = rad * __cosf(theta) * std0 + mu0;
            float ov1  = rad * __sinf(theta) * std1 + mu1;
            out[g0] = ov0;
            out[g1] = ov1;
            part += 1.0f + lv0 - ov0 * ov0 - std0 * std0;
            part += 1.0f + lv1 - ov1 * ov1 - std1 * std1;
        }
    }
    part *= -0.5f;

    // wave32 reduction + one atomic per wave
    #pragma unroll
    for (int off = 16; off > 0; off >>= 1)
        part += __shfl_xor(part, off, 32);
    if (lane == 0)
        atomicAdd(loss, part);
}

extern "C" void kernel_launch(void* const* d_in, const int* in_sizes, int n_in,
                              void* d_out, int out_size, void* d_ws, size_t ws_size,
                              hipStream_t stream) {
    const float* x   = (const float*)d_in[0];
    const float* wmu = (const float*)d_in[1];
    const float* bmu = (const float*)d_in[2];
    const float* wlv = (const float*)d_in[3];
    const float* blv = (const float*)d_in[4];

    float* out  = (float*)d_out;
    float* loss = out + (size_t)BIMG * OOUT * HWSZ;   // scalar tail of d_out
    unsigned short* wbf = (unsigned short*)d_ws;      // 2*73728*2 = 294912 B

    // prologue: bf16 weight conversion + loss zeroing (graph-capture safe)
    convert_weights_kernel<<<dim3((WELEM + 255) / 256), dim3(256), 0, stream>>>(
        wmu, wlv, wbf, loss);

    dim3 grid(HWSZ / NTILE, BIMG);   // 128 x 16 workgroups
    varconv_wmma_kernel<<<grid, dim3(256), 0, stream>>>(x, wbf, bmu, blv,
                                                        out, loss);
}
